// MultiHeadAttention_70489003262200
// MI455X (gfx1250) — matrix operand order repaired and checked by reading, not compiled, not run on hardware
//
#include <hip/hip_runtime.h>

// ---------------- problem constants ----------------
#define BB 2
#define SS 2048
#define DD 1024
#define HH 16
#define DKK 64

typedef __attribute__((ext_vector_type(16))) __bf16          v16bf;
typedef __attribute__((ext_vector_type(8)))  float           v8f;
typedef __attribute__((ext_vector_type(16))) unsigned short  v16u;
typedef __attribute__((ext_vector_type(8)))  unsigned short  v8u;
typedef __attribute__((ext_vector_type(4)))  unsigned short  v4u;

// fp32 -> bf16 round-to-nearest-even
static __device__ __forceinline__ unsigned short bf16_rn(float f) {
    unsigned int x = __builtin_bit_cast(unsigned int, f);
    unsigned int r = x + 0x7FFFu + ((x >> 16) & 1u);
    return (unsigned short)(r >> 16);
}

static __device__ __forceinline__ v8f wmma_bf16(v16bf a, v16bf b, v8f c) {
    return __builtin_amdgcn_wmma_f32_16x16x32_bf16(false, a, false, b, (short)0, c, false, false);
}

// ---- A fragment (16x32 bf16), ISA 7.12.2:
// lane l=lane&15 holds row M=row0+l; hi=lane>>4 selects K sub-block.
// elems 0..7 = K k0+hi*8 .. +7 ; elems 8..15 = K k0+16+hi*8 .. +7
static __device__ __forceinline__ v16bf load_a_bf16(const unsigned short* src, int ld,
                                                    int row0, int k0, int lane) {
    int l = lane & 15, hi = lane >> 4;
    const unsigned short* p = src + (size_t)(row0 + l) * ld;
    v8u lo = *(const v8u*)(p + k0 + hi * 8);        // 16B aligned
    v8u h2 = *(const v8u*)(p + k0 + 16 + hi * 8);   // 16B aligned
    v16u u;
#pragma unroll
    for (int i = 0; i < 8; ++i) { u[i] = lo[i]; u[8 + i] = h2[i]; }
    return __builtin_bit_cast(v16bf, u);
}

// ---- B fragment (32x16 bf16): B[kk][n] = src[n][kk], src row-major with ld.
// lane l holds column n=n0+l; elems 0..7 = K k0+hi*8 .. +7 ; elems 8..15 = K k0+16+hi*8 .. +7
static __device__ __forceinline__ v16bf load_b_bf16(const unsigned short* src, int ld,
                                                    int n0, int k0, int lane) {
    int l = lane & 15, hi = lane >> 4;
    const unsigned short* p = src + (size_t)(n0 + l) * ld + k0 + hi * 8;
    v8u lo = *(const v8u*)p;                        // 16B aligned
    v8u h2 = *(const v8u*)(p + 16);                 // 16B aligned
    v16u u;
#pragma unroll
    for (int i = 0; i < 8; ++i) { u[i] = lo[i]; u[8 + i] = h2[i]; }
    return __builtin_bit_cast(v16bf, u);
}

// =====================================================================
// One-shot fp32 -> bf16 conversion (bandwidth-bound; keeps conversion
// VALU out of the WMMA hot loops).
// =====================================================================
__global__ void __launch_bounds__(256)
cvt_f32_bf16_kernel(const float* __restrict__ src, unsigned short* __restrict__ dst, int n4) {
    const int i = blockIdx.x * 256 + threadIdx.x;
    if (i < n4) {
        const float4 f = ((const float4*)src)[i];
        v4u u;
        u[0] = bf16_rn(f.x); u[1] = bf16_rn(f.y);
        u[2] = bf16_rn(f.z); u[3] = bf16_rn(f.w);
        ((v4u*)dst)[i] = u;
    }
}

// =====================================================================
// GEMM: out[m][n] = sum_k A[m][k] * W[n][k] + bias[n]     (A @ W^T + b)
// A and W both pre-converted bf16. 256 threads = 8 waves (4 M x 2 N);
// block tile 128x128; wave tile 32x64 = 8 accumulators; K step 32
// -> 8 v_wmma per step from 2 A-frags + 4 B-frags (pure b128 loads).
// OUTMODE 0: bf16 dst[B][H][S][DK]    (split heads)
// OUTMODE 1: bf16 dst[B][H][DK][S]    (split heads, transposed: for V)
// OUTMODE 2: f32  dst[m][n] row-major (final output)
// =====================================================================
template <int OUTMODE>
__global__ void __launch_bounds__(256)
gemm_bf16_kernel(const unsigned short* __restrict__ A, const unsigned short* __restrict__ W,
                 const float* __restrict__ bias, void* __restrict__ dst, int Kdim) {
    const int lane = threadIdx.x & 31;
    const int wave = threadIdx.x >> 5;
    const int wm = wave & 3, wn = wave >> 2;
    const int m0 = blockIdx.x * 128 + wm * 32;
    const int n0 = blockIdx.y * 128 + wn * 64;

    v8f acc[2][4] = {};
    for (int k = 0; k < Kdim; k += 32) {
        const v16bf a0 = load_a_bf16(A, Kdim, m0, k, lane);
        const v16bf a1 = load_a_bf16(A, Kdim, m0 + 16, k, lane);
        v16bf b[4];
#pragma unroll
        for (int j = 0; j < 4; ++j) b[j] = load_b_bf16(W, Kdim, n0 + j * 16, k, lane);
#pragma unroll
        for (int j = 0; j < 4; ++j) {
            acc[0][j] = wmma_bf16(a0, b[j], acc[0][j]);
            acc[1][j] = wmma_bf16(a1, b[j], acc[1][j]);
        }
    }

    const int l = lane & 15, hi = lane >> 4;
#pragma unroll
    for (int i = 0; i < 2; ++i)
#pragma unroll
        for (int j = 0; j < 4; ++j) {
            const int col = n0 + j * 16 + l;
            const float bv = bias[col];
#pragma unroll
            for (int r = 0; r < 8; ++r) {
                const int row = m0 + i * 16 + hi * 8 + r;
                const float v = acc[i][j][r] + bv;
                if constexpr (OUTMODE == 2) {
                    ((float*)dst)[(size_t)row * DD + col] = v;
                } else {
                    const unsigned short uv = bf16_rn(v);
                    const int b2 = row >> 11, s = row & (SS - 1);   // S = 2048
                    const int h  = col >> 6,  dk = col & (DKK - 1); // DK = 64
                    if constexpr (OUTMODE == 0)
                        ((unsigned short*)dst)[(((size_t)(b2 * HH + h)) * SS + s) * DKK + dk] = uv;
                    else
                        ((unsigned short*)dst)[(((size_t)(b2 * HH + h)) * DKK + dk) * SS + s] = uv;
                }
            }
        }
}

// =====================================================================
// Causal flash attention. Grid: (S/64, B*H). 128 threads = 4 waves,
// wave w owns query rows [qt*64 + w*16, +16). Q fragments in registers.
// Per 64-key block:
//   - K tile (64x64) and V^T tile (64x64) staged ONCE per workgroup
//     into LDS with cooperative b128 copies (4x cut in L2 read traffic
//     vs per-wave fragment loads),
//   - 8 WMMA scores + 8 WMMA P@V, fragments served from LDS,
//   - one exp2-domain online-softmax pass (shfl_xor reductions within
//     16-lane halves = C-layout rows),
//   - P re-fragmented via per-wave LDS region.
// Next K/V tiles prefetched (global_prefetch_b8) to warm L2.
// =====================================================================
__global__ void __launch_bounds__(128)
attn_kernel(const unsigned short* __restrict__ qbuf,
            const unsigned short* __restrict__ kbuf,
            const unsigned short* __restrict__ vtbuf,
            unsigned short* __restrict__ obuf) {
    __shared__ __attribute__((aligned(32))) unsigned short kt[64 * 64];       // [key][dk]
    __shared__ __attribute__((aligned(32))) unsigned short vtile[64 * 64];    // [dk][key]
    __shared__ __attribute__((aligned(32))) unsigned short pshared[4][16 * 64];

    const int lane = threadIdx.x & 31;
    const int wave = threadIdx.x >> 5;
    const int qt = blockIdx.x;   // query tile: 64 rows
    const int bh = blockIdx.y;   // b*H + h

    const unsigned short* q  = qbuf  + (size_t)bh * SS * DKK;
    const unsigned short* kp = kbuf  + (size_t)bh * SS * DKK;
    const unsigned short* vt = vtbuf + (size_t)bh * DKK * SS;

    const int qrow0 = qt * 64 + wave * 16;
    const v16bf qa0 = load_a_bf16(q, DKK, qrow0, 0, lane);
    const v16bf qa1 = load_a_bf16(q, DKK, qrow0, 32, lane);

    v8f o[4] = {};
    float mrow[8], lrow[8];
#pragma unroll
    for (int r = 0; r < 8; ++r) { mrow[r] = -3.0e38f; lrow[r] = 0.0f; }

    const int l = lane & 15, hi = lane >> 4;
    const int srow  = threadIdx.x >> 1;          // 0..63 : staging row
    const int shalf = (threadIdx.x & 1) * 32;    // 0/32  : staging half-row (ushorts)
    const int jend = qt * 64 + 64;               // causal: uniform across block
    const float kscale = 0.18033688011112042f;   // (1/sqrt(64)) * log2(e)

    for (int jb = 0; jb < jend; jb += 64) {
        // prefetch next 64-key K tile and V^T tile (warm L2 for next staging)
        if (jb + 64 < jend) {
            __builtin_prefetch(kp + (size_t)(jb + 64 + lane) * DKK, 0, 1);
            __builtin_prefetch(kp + (size_t)(jb + 96 + lane) * DKK, 0, 1);
            __builtin_prefetch(vt + (size_t)(2 * lane) * SS + jb + 64, 0, 1);
            __builtin_prefetch(vt + (size_t)(2 * lane + 1) * SS + jb + 64, 0, 1);
        }

        __syncthreads();   // previous iteration's LDS reads complete
        // ---- cooperative staging: K tile + V^T tile -> LDS (b128 copies)
#pragma unroll
        for (int i = 0; i < 4; ++i)
            *(uint4*)(kt + srow * 64 + shalf + i * 8) =
                *(const uint4*)(kp + (size_t)(jb + srow) * DKK + shalf + i * 8);
#pragma unroll
        for (int i = 0; i < 4; ++i)
            *(uint4*)(vtile + srow * 64 + shalf + i * 8) =
                *(const uint4*)(vt + (size_t)srow * SS + jb + shalf + i * 8);
        __syncthreads();   // tiles visible to all waves

        // ---- scores: (16x64) @ (64x64) -> four 16x16 f32 tiles (K from LDS)
        v8f sc[4];
#pragma unroll
        for (int ns = 0; ns < 4; ++ns) {
            v8f s = {};
            v16bf b0 = load_b_bf16(kt, 64, ns * 16, 0, lane);
            v16bf b1 = load_b_bf16(kt, 64, ns * 16, 32, lane);
            s = wmma_bf16(qa0, b0, s);
            s = wmma_bf16(qa1, b1, s);
            sc[ns] = s;
        }

        // ---- scale to log2 domain, causal mask, online softmax (one pass / 64 keys)
#pragma unroll
        for (int r = 0; r < 8; ++r) {
            const int row = qrow0 + hi * 8 + r;
#pragma unroll
            for (int ns = 0; ns < 4; ++ns) {
                const int col = jb + ns * 16 + l;
                const float v = sc[ns][r] * kscale;
                sc[ns][r] = (col <= row) ? v : -3.0e38f;
            }
            float mx = fmaxf(fmaxf(sc[0][r], sc[1][r]), fmaxf(sc[2][r], sc[3][r]));
#pragma unroll
            for (int off = 1; off < 16; off <<= 1)
                mx = fmaxf(mx, __shfl_xor(mx, off, 32));
            const float mnew = fmaxf(mrow[r], mx);
            const float rescale = exp2f(mrow[r] - mnew);
            mrow[r] = mnew;
            float psum = 0.0f;
#pragma unroll
            for (int ns = 0; ns < 4; ++ns) {
                const float sv = sc[ns][r];
                const float p = (sv > -1.0e37f) ? exp2f(sv - mnew) : 0.0f;
                sc[ns][r] = p;
                psum += p;
            }
#pragma unroll
            for (int off = 1; off < 16; off <<= 1)
                psum += __shfl_xor(psum, off, 32);
            lrow[r] = lrow[r] * rescale + psum;
#pragma unroll
            for (int d = 0; d < 4; ++d) o[d][r] *= rescale;
        }

        // ---- re-fragment P (C layout -> A layout, 16x64) through per-wave LDS
        unsigned short* pw = pshared[wave];
#pragma unroll
        for (int ns = 0; ns < 4; ++ns)
#pragma unroll
            for (int r = 0; r < 8; ++r)
                pw[(hi * 8 + r) * 64 + ns * 16 + l] = bf16_rn(sc[ns][r]);
        __syncthreads();   // P visible (uniform control flow)
        const v16bf pa0 = load_a_bf16(pw, 64, 0, 0, lane);
        const v16bf pa1 = load_a_bf16(pw, 64, 0, 32, lane);

        // ---- O += P @ V   (V^T tile from LDS: contiguous B fragments)
#pragma unroll
        for (int d = 0; d < 4; ++d) {
            v16bf bv0 = load_b_bf16(vtile, 64, d * 16, 0, lane);
            v16bf bv1 = load_b_bf16(vtile, 64, d * 16, 32, lane);
            o[d] = wmma_bf16(pa0, bv0, o[d]);
            o[d] = wmma_bf16(pa1, bv1, o[d]);
        }
    }

    // ---- normalize + store attn output as bf16 [B][S][D] for the O-proj GEMM
    const int b = bh >> 4, h = bh & (HH - 1);
#pragma unroll
    for (int r = 0; r < 8; ++r) {
        const float invl = 1.0f / lrow[r];
        const int srowq = qrow0 + hi * 8 + r;
        const size_t base = ((size_t)b * SS + srowq) * DD + h * DKK;
#pragma unroll
        for (int d = 0; d < 4; ++d)
            obuf[base + d * 16 + l] = bf16_rn(o[d][r] * invl);
    }
}

// =====================================================================
extern "C" void kernel_launch(void* const* d_in, const int* in_sizes, int n_in,
                              void* d_out, int out_size, void* d_ws, size_t ws_size,
                              hipStream_t stream) {
    (void)in_sizes; (void)n_in; (void)out_size; (void)ws_size;

    const float* Q  = (const float*)d_in[0];
    const float* K  = (const float*)d_in[1];
    const float* V  = (const float*)d_in[2];
    /* d_in[3] = mask (causal; applied analytically) */
    const float* Wq = (const float*)d_in[4];
    const float* bq = (const float*)d_in[5];
    const float* Wk = (const float*)d_in[6];
    const float* bk = (const float*)d_in[7];
    const float* Wv = (const float*)d_in[8];
    const float* bv = (const float*)d_in[9];
    const float* Wo = (const float*)d_in[10];
    const float* bo = (const float*)d_in[11];

    // workspace carve-up (all bf16 / ushort)
    const size_t act  = (size_t)BB * SS * DD;   // 4 Mi elements
    const size_t wsz  = (size_t)DD * DD;        // 1 Mi elements
    unsigned short* ws_q    = (unsigned short*)d_ws;     // post-proj q  [B][H][S][DK]
    unsigned short* ws_k    = ws_q    + act;             // post-proj k  [B][H][S][DK]
    unsigned short* ws_vt   = ws_k    + act;             // post-proj v  [B][H][DK][S]
    unsigned short* ws_attn = ws_vt   + act;             // attn out     [B][S][D]
    unsigned short* ws_Qb   = ws_attn + act;             // bf16 copies of inputs
    unsigned short* ws_Kb   = ws_Qb   + act;
    unsigned short* ws_Vb   = ws_Kb   + act;
    unsigned short* ws_Wq   = ws_Vb   + act;
    unsigned short* ws_Wk   = ws_Wq   + wsz;
    unsigned short* ws_Wv   = ws_Wk   + wsz;
    unsigned short* ws_Wo   = ws_Wv   + wsz;

    // ---- stage 0: one-shot fp32 -> bf16 conversions (bandwidth-bound)
    {
        const int a4 = (int)(act / 4), w4 = (int)(wsz / 4);
        cvt_f32_bf16_kernel<<<(a4 + 255) / 256, 256, 0, stream>>>(Q,  ws_Qb, a4);
        cvt_f32_bf16_kernel<<<(a4 + 255) / 256, 256, 0, stream>>>(K,  ws_Kb, a4);
        cvt_f32_bf16_kernel<<<(a4 + 255) / 256, 256, 0, stream>>>(V,  ws_Vb, a4);
        cvt_f32_bf16_kernel<<<(w4 + 255) / 256, 256, 0, stream>>>(Wq, ws_Wq, w4);
        cvt_f32_bf16_kernel<<<(w4 + 255) / 256, 256, 0, stream>>>(Wk, ws_Wk, w4);
        cvt_f32_bf16_kernel<<<(w4 + 255) / 256, 256, 0, stream>>>(Wv, ws_Wv, w4);
        cvt_f32_bf16_kernel<<<(w4 + 255) / 256, 256, 0, stream>>>(Wo, ws_Wo, w4);
    }

    // ---- stage 1: QKV projections (pure-WMMA GEMMs)
    const dim3 ggrid((BB * SS) / 128, DD / 128);        // (32, 8)
    gemm_bf16_kernel<0><<<ggrid, 256, 0, stream>>>(ws_Qb, ws_Wq, bq, ws_q,  DD);
    gemm_bf16_kernel<0><<<ggrid, 256, 0, stream>>>(ws_Kb, ws_Wk, bk, ws_k,  DD);
    gemm_bf16_kernel<1><<<ggrid, 256, 0, stream>>>(ws_Vb, ws_Wv, bv, ws_vt, DD);

    // ---- stage 2: causal flash attention (64-key blocks, LDS-staged K/V)
    attn_kernel<<<dim3(SS / 64, BB * HH), 128, 0, stream>>>(ws_q, ws_k, ws_vt, ws_attn);

    // ---- stage 3: output projection (fp32 out + bias)
    gemm_bf16_kernel<2><<<ggrid, 256, 0, stream>>>(ws_attn, ws_Wo, bo, d_out, DD);
}
